// ValueNetwork_51324859187768
// MI455X (gfx1250) — matrix operand order repaired and checked by reading, not compiled, not run on hardware
//
#include <hip/hip_runtime.h>

// ---------------------------------------------------------------------------
// Fully fused ValueNetwork forward for MI455X (gfx1250, wave32, WMMA).
//
// Math folding:  h1 = relu( e_h @ (c1rh_root + c1hh_root - c1hh_rel/95)
//                           + [e_r@c1rh_rel + (sum_n e_h)@c1hh_rel/95
//                              + c1rh_b + c1hh_b] )
// so the three per-row (B*N) GEMMs collapse into ONE K=64 WMMA GEMM.
// One wave (32 threads) handles one batch element end-to-end; the only HBM
// traffic is the 20.4MB state tensor (+B floats out); weights are
// pre-swizzled once into f16 WMMA B-fragments in d_ws and stay L2-resident.
//
// v2: column-sum partials kept in registers across all 6 row-tiles (the
// column a lane covers is tile-invariant), reduced across lane-halves ONCE
// per pass -> removes ~46 barrier+LDS round-trips from the WMMA inner loop.
// Next-tile state rows prefetched (global_prefetch_b8).
// ---------------------------------------------------------------------------

typedef _Float16 v16h __attribute__((ext_vector_type(16)));
typedef float    v8f  __attribute__((ext_vector_type(8)));
typedef unsigned int   u32;
typedef unsigned short u16;

union Frag { v16h h; u32 u[8]; uint4 q[2]; };

#define WMMA_F16(a, b, c) \
  __builtin_amdgcn_wmma_f32_16x16x32_f16(false, (a), false, (b), (short)0, (c), false, false)

__device__ __forceinline__ u32 pk2(float a, float b) {
  u16 ua = __builtin_bit_cast(u16, (_Float16)a);
  u16 ub = __builtin_bit_cast(u16, (_Float16)b);
  return (u32)ua | ((u32)ub << 16);
}
__device__ __forceinline__ u16 f16b(float a) {
  return __builtin_bit_cast(u16, (_Float16)a);
}
// A-fragment K index for 16x32 f16 A (contiguous, matches B frags): VGPR v, lane-half hf
__device__ __forceinline__ int kA(int v, int hf) {
  return hf * 16 + (v << 1);
}

// ---------------------------------------------------------------------------
// Prep: build f16 pre-swizzled B-fragments in workspace (u32 layout):
//   [0,1024)    WH1 : (K=7 padded to 32) x 64, frags [nt][lane][v]
//   [1024,3072) WH2 : 64 x 64,  frags [kt][nt][lane][v]
//   [3072,5120) M1  : 64 x (50 padded to 64), combined h1 matrix
// B-fragment mapping (32x16 f16): lane lo = col n, half hf: K = hf*16 + 2v.
// ---------------------------------------------------------------------------
__global__ void prep_weights(const float* __restrict__ wh_w1,
                             const float* __restrict__ wh_w2,
                             const float* __restrict__ c1rh_root,
                             const float* __restrict__ c1hh_root,
                             const float* __restrict__ c1hh_rel,
                             u32* __restrict__ ws) {
  const float inv95 = 1.0f / 95.0f;
  int i = blockIdx.x * blockDim.x + threadIdx.x;
  if (i >= 5120) return;
  if (i < 1024) {                               // WH1 (zero-padded K 7->32)
    int nt = i >> 8, lane = (i >> 3) & 31, v = i & 7;
    int lo = lane & 15, hf = lane >> 4;
    int k = hf * 16 + v * 2;
    int n = nt * 16 + lo;
    float a = (k     < 7) ? wh_w1[k * 64 + n]       : 0.f;
    float b = (k + 1 < 7) ? wh_w1[(k + 1) * 64 + n] : 0.f;
    ws[i] = pk2(a, b);
  } else if (i < 3072) {                        // WH2 64x64
    int j = i - 1024;
    int kt = j >> 10, nt = (j >> 8) & 3, lane = (j >> 3) & 31, v = j & 7;
    int lo = lane & 15, hf = lane >> 4;
    int k = kt * 32 + hf * 16 + v * 2;
    int n = nt * 16 + lo;
    ws[i] = pk2(wh_w2[k * 64 + n], wh_w2[(k + 1) * 64 + n]);
  } else {                                      // M1 = root_rh + root_hh - rel_hh/95
    int j = i - 3072;
    int kt = j >> 10, nt = (j >> 8) & 3, lane = (j >> 3) & 31, v = j & 7;
    int lo = lane & 15, hf = lane >> 4;
    int k = kt * 32 + hf * 16 + v * 2;
    int n = nt * 16 + lo;
    float a = 0.f, b = 0.f;
    if (n < 50) {
      a = c1rh_root[k * 50 + n] + c1hh_root[k * 50 + n] - c1hh_rel[k * 50 + n] * inv95;
      b = c1rh_root[(k + 1) * 50 + n] + c1hh_root[(k + 1) * 50 + n]
          - c1hh_rel[(k + 1) * 50 + n] * inv95;
    }
    ws[i] = pk2(a, b);
  }
}

// ---------------------------------------------------------------------------
// Main fused kernel: 1 block = 1 wave = 1 batch element.
// ---------------------------------------------------------------------------
__global__ void __launch_bounds__(32)
value_net_fused(const float* __restrict__ state,
                const float* __restrict__ wr_w1, const float* __restrict__ wr_b1,
                const float* __restrict__ wr_w2, const float* __restrict__ wr_b2,
                const float* __restrict__ wh_b1, const float* __restrict__ wh_b2,
                const float* __restrict__ c1hr_rel, const float* __restrict__ c1hr_root,
                const float* __restrict__ c1hr_b,
                const float* __restrict__ c1rh_rel, const float* __restrict__ c1rh_b,
                const float* __restrict__ c1hh_rel, const float* __restrict__ c1hh_b,
                const float* __restrict__ c2hr_rel, const float* __restrict__ c2hr_root,
                const float* __restrict__ c2hr_b,
                const float* __restrict__ vn_w1, const float* __restrict__ vn_b1,
                const float* __restrict__ vn_w2, const float* __restrict__ vn_b2,
                const float* __restrict__ vn_w3, const float* __restrict__ vn_b3,
                const float* __restrict__ vn_w4, const float* __restrict__ vn_b4,
                const u32* __restrict__ wsf,
                float* __restrict__ out) {
  __shared__ u16   ehlds[96][64];   // e_h tile store (f16 bits), 12 KB
  __shared__ u16   htile[16][64];   // layer-1 hidden tile (f16 bits), 2 KB
  __shared__ float er[64], sumh[64], vvec[64], r1v[64], h1m[64];
  __shared__ float m1v[160], m2v[112], m3v[112], scr[64];
  __shared__ float red4[4][32];     // cross-half reduction scratch

  const int lane = threadIdx.x;
  const int lo = lane & 15, hf = lane >> 4;
  const int b = blockIdx.x;
  const float* st = state + (size_t)b * 96 * 13;
  const v8f zero = {};

  // ---------------- e_r : relu(relu(ss@wr_w1+b1)@wr_w2+b2) (VALU, tiny) ----
  float ss0[6];
#pragma unroll
  for (int k = 0; k < 6; k++) ss0[k] = st[k];
#pragma unroll
  for (int h = 0; h < 2; h++) {
    int c = lane + h * 32;
    float acc = wr_b1[c];
#pragma unroll
    for (int k = 0; k < 6; k++) acc += ss0[k] * wr_w1[k * 64 + c];
    scr[c] = fmaxf(acc, 0.f);
  }
  __syncthreads();
#pragma unroll
  for (int h = 0; h < 2; h++) {
    int c = lane + h * 32;
    float acc = wr_b2[c];
    for (int k = 0; k < 64; k++) acc += scr[k] * wr_w2[k * 64 + c];
    er[c] = fmaxf(acc, 0.f);
  }
  __syncthreads();

  // ---------------- load pre-swizzled WMMA B fragments ---------------------
  const uint4* wq = (const uint4*)wsf;
  Frag wh1f[4], wh2f[2][4];
#pragma unroll
  for (int nt = 0; nt < 4; nt++) {
    wh1f[nt].q[0] = wq[nt * 64 + lane * 2];
    wh1f[nt].q[1] = wq[nt * 64 + lane * 2 + 1];
  }
#pragma unroll
  for (int kt = 0; kt < 2; kt++)
#pragma unroll
    for (int nt = 0; nt < 4; nt++) {
      wh2f[kt][nt].q[0] = wq[256 + (kt * 4 + nt) * 64 + lane * 2];
      wh2f[kt][nt].q[1] = wq[256 + (kt * 4 + nt) * 64 + lane * 2 + 1];
    }

  // ---------------- pass 1: e_h = MLP(7->64->64), keep f16 + column sums ---
  // Lane's output column (nt*16+lo) is tile-invariant -> keep column-sum
  // partials in registers across all 6 tiles; reduce across halves once.
  float sacc[4] = {0.f, 0.f, 0.f, 0.f};
  for (int t = 0; t < 6; t++) {
    const int n0 = t * 16;
    // prefetch next tile's rows (clamped; speculative, dropped if redundant)
    {
      int pr = n0 + 16 + lo; pr = pr < 96 ? pr : 95;
      __builtin_prefetch(st + (size_t)pr * 13, 0, 1);
    }
    // layer-1 A fragment: row m = lo, K = 7 real values zero-padded to 32
    const float* row = st + (size_t)(n0 + lo) * 13 + 6;
    float x[8];
#pragma unroll
    for (int j = 0; j < 7; j++) x[j] = row[j];
    x[7] = 0.f;
    Frag a1;
#pragma unroll
    for (int v = 0; v < 8; v++)
      a1.u[v] = (hf == 0 && v < 4) ? pk2(x[v * 2], x[v * 2 + 1]) : 0u;

    v8f acc1[4];
#pragma unroll
    for (int nt = 0; nt < 4; nt++) acc1[nt] = WMMA_F16(a1.h, wh1f[nt].h, zero);

    __syncthreads();                       // htile from prev iter fully read
#pragma unroll
    for (int nt = 0; nt < 4; nt++) {
      int col = nt * 16 + lo;
      float bb = wh_b1[col];
#pragma unroll
      for (int v = 0; v < 8; v++)
        htile[v + 8 * hf][col] = f16b(fmaxf(acc1[nt][v] + bb, 0.f));
    }
    __syncthreads();

    // layer-2 A fragments straight out of LDS (adjacent f16 pair = 1 dword)
    Frag a20, a21;
#pragma unroll
    for (int v = 0; v < 8; v++) {
      int k = kA(v, hf);
      a20.u[v] = *(const u32*)&htile[lo][k];
      a21.u[v] = *(const u32*)&htile[lo][32 + k];
    }
#pragma unroll
    for (int nt = 0; nt < 4; nt++) {
      v8f d = WMMA_F16(a20.h, wh2f[0][nt].h, zero);
      d     = WMMA_F16(a21.h, wh2f[1][nt].h, d);
      int col = nt * 16 + lo;
      float bb = wh_b2[col];
      float cp = 0.f;
#pragma unroll
      for (int v = 0; v < 8; v++) {
        float eh = fmaxf(d[v] + bb, 0.f);
        ehlds[n0 + v + 8 * hf][col] = f16b(eh);
        cp += eh;
      }
      sacc[nt] += cp;                      // register-resident partial
    }
  }
  __syncthreads();
#pragma unroll
  for (int nt = 0; nt < 4; nt++) red4[nt][lane] = sacc[nt];
  __syncthreads();
  if (hf == 0)
#pragma unroll
    for (int nt = 0; nt < 4; nt++)
      sumh[nt * 16 + lo] = red4[nt][lo] + red4[nt][lo + 16];
  __syncthreads();

  // ---------------- v_b and r1 (per-b GEMVs, VALU) -------------------------
  for (int c = lane; c < 50; c += 32) {
    float a = c1rh_b[c] + c1hh_b[c];
    float r = c1hr_b[c];
    for (int k = 0; k < 64; k++) {
      float e = er[k], s = sumh[k];
      a += e * c1rh_rel[k * 50 + c] + (s * (1.f / 95.f)) * c1hh_rel[k * 50 + c];
      r += e * c1hr_root[k * 50 + c] + (s * (1.f / 96.f)) * c1hr_rel[k * 50 + c];
    }
    vvec[c] = a;
    r1v[c] = fmaxf(r, 0.f);
  }
  if (lane < 14) vvec[50 + lane] = 0.f;
  __syncthreads();

  // ---------------- pass 2: h1 = relu(e_h @ M1 + v_b), mean over n ---------
  Frag m1f[2][4];
#pragma unroll
  for (int kt = 0; kt < 2; kt++)
#pragma unroll
    for (int nt = 0; nt < 4; nt++) {
      m1f[kt][nt].q[0] = wq[768 + (kt * 4 + nt) * 64 + lane * 2];
      m1f[kt][nt].q[1] = wq[768 + (kt * 4 + nt) * 64 + lane * 2 + 1];
    }
  float hacc[4] = {0.f, 0.f, 0.f, 0.f};
  for (int t = 0; t < 6; t++) {
    const int n0 = t * 16;
    Frag a0, a1f;
#pragma unroll
    for (int v = 0; v < 8; v++) {
      int k = kA(v, hf);
      a0.u[v]  = *(const u32*)&ehlds[n0 + lo][k];
      a1f.u[v] = *(const u32*)&ehlds[n0 + lo][32 + k];
    }
#pragma unroll
    for (int nt = 0; nt < 4; nt++) {
      v8f d = WMMA_F16(a0.h, m1f[0][nt].h, zero);
      d     = WMMA_F16(a1f.h, m1f[1][nt].h, d);
      int col = nt * 16 + lo;
      float vv = vvec[col];
      float cp = 0.f;
#pragma unroll
      for (int v = 0; v < 8; v++)
        cp += (col < 50) ? fmaxf(d[v] + vv, 0.f) : 0.f;
      hacc[nt] += cp;                      // register-resident partial
    }
  }
  __syncthreads();
#pragma unroll
  for (int nt = 0; nt < 4; nt++) red4[nt][lane] = hacc[nt];
  __syncthreads();
  if (hf == 0)
#pragma unroll
    for (int nt = 0; nt < 4; nt++)
      h1m[nt * 16 + lo] = (red4[nt][lo] + red4[nt][lo + 16]) * (1.f / 96.f);
  __syncthreads();

  // ---------------- r2 and value-MLP 32->150->100->100->1 ------------------
  {
    float acc = c2hr_b[lane];
    for (int k = 0; k < 50; k++)
      acc += r1v[k] * c2hr_root[k * 32 + lane] + h1m[k] * c2hr_rel[k * 32 + lane];
    scr[lane] = fmaxf(acc, 0.f);           // r2 in scr[0..31]
  }
  __syncthreads();
  for (int c = lane; c < 150; c += 32) {
    float acc = vn_b1[c];
#pragma unroll
    for (int k = 0; k < 32; k++) acc += scr[k] * vn_w1[k * 150 + c];
    m1v[c] = fmaxf(acc, 0.f);
  }
  __syncthreads();
  for (int c = lane; c < 100; c += 32) {
    float acc = vn_b2[c];
    for (int k = 0; k < 150; k++) acc += m1v[k] * vn_w2[k * 100 + c];
    m2v[c] = fmaxf(acc, 0.f);
  }
  __syncthreads();
  for (int c = lane; c < 100; c += 32) {
    float acc = vn_b3[c];
    for (int k = 0; k < 100; k++) acc += m2v[k] * vn_w3[k * 100 + c];
    m3v[c] = fmaxf(acc, 0.f);
  }
  __syncthreads();
  float p = 0.f;
  for (int k = lane; k < 100; k += 32) p += m3v[k] * vn_w4[k];
  red4[0][lane] = p;
  __syncthreads();
  if (lane < 16) red4[0][lane] += red4[0][lane + 16];
  __syncthreads();
  if (lane < 8) red4[0][lane] += red4[0][lane + 8];
  __syncthreads();
  if (lane < 4) red4[0][lane] += red4[0][lane + 4];
  __syncthreads();
  if (lane < 2) red4[0][lane] += red4[0][lane + 2];
  __syncthreads();
  if (lane == 0) out[b] = red4[0][0] + red4[0][1] + vn_b4[0];
}

// ---------------------------------------------------------------------------
extern "C" void kernel_launch(void* const* d_in, const int* in_sizes, int n_in,
                              void* d_out, int out_size, void* d_ws, size_t ws_size,
                              hipStream_t stream) {
  (void)n_in; (void)out_size; (void)ws_size;
  const float* state     = (const float*)d_in[0];
  const float* wr_w1     = (const float*)d_in[1];
  const float* wr_b1     = (const float*)d_in[2];
  const float* wr_w2     = (const float*)d_in[3];
  const float* wr_b2     = (const float*)d_in[4];
  const float* wh_w1     = (const float*)d_in[5];
  const float* wh_b1     = (const float*)d_in[6];
  const float* wh_w2     = (const float*)d_in[7];
  const float* wh_b2     = (const float*)d_in[8];
  const float* c1hr_rel  = (const float*)d_in[9];
  const float* c1hr_root = (const float*)d_in[10];
  const float* c1hr_b    = (const float*)d_in[11];
  const float* c1rh_rel  = (const float*)d_in[12];
  const float* c1rh_root = (const float*)d_in[13];
  const float* c1rh_b    = (const float*)d_in[14];
  const float* c1hh_rel  = (const float*)d_in[15];
  const float* c1hh_root = (const float*)d_in[16];
  const float* c1hh_b    = (const float*)d_in[17];
  const float* c2hr_rel  = (const float*)d_in[18];
  const float* c2hr_root = (const float*)d_in[19];
  const float* c2hr_b    = (const float*)d_in[20];
  const float* vn_w1     = (const float*)d_in[21];
  const float* vn_b1     = (const float*)d_in[22];
  const float* vn_w2     = (const float*)d_in[23];
  const float* vn_b2     = (const float*)d_in[24];
  const float* vn_w3     = (const float*)d_in[25];
  const float* vn_b3     = (const float*)d_in[26];
  const float* vn_w4     = (const float*)d_in[27];
  const float* vn_b4     = (const float*)d_in[28];

  u32* ws = (u32*)d_ws;
  const int B = in_sizes[0] / (96 * 13);

  prep_weights<<<20, 256, 0, stream>>>(wh_w1, wh_w2, c1rh_root, c1hh_root, c1hh_rel, ws);
  value_net_fused<<<B, 32, 0, stream>>>(
      state, wr_w1, wr_b1, wr_w2, wr_b2, wh_b1, wh_b2,
      c1hr_rel, c1hr_root, c1hr_b, c1rh_rel, c1rh_b, c1hh_rel, c1hh_b,
      c2hr_rel, c2hr_root, c2hr_b,
      vn_w1, vn_b1, vn_w2, vn_b2, vn_w3, vn_b3, vn_w4, vn_b4,
      ws, (float*)d_out);
}
